// UnifiedRetrievalTransfer_40200893890564
// MI455X (gfx1250) — compile-verified, matrix operand order checked by reading, not run on hardware
//
#include <hip/hip_runtime.h>
#include <hip/hip_bf16.h>
#include <math.h>
#include <stdint.h>

// ---------------- constants ----------------
#define BATCH 4096
#define KNBR  32
#define NS    6
#define QD    512
#define HDIM  384
#define NH    6
#define HD2   192   // H/2
#define NRT   21    // NR+1

typedef __attribute__((ext_vector_type(16))) _Float16 v16h;
typedef __attribute__((ext_vector_type(8)))  _Float16 v8h;
typedef __attribute__((ext_vector_type(8)))  float    v8f;

// ---------------- device helpers ----------------
__device__ __forceinline__ float geluf(float x) {
    return 0.5f * x * (1.0f + erff(x * 0.7071067811865475f));
}
__device__ __forceinline__ v8f vzero8() {
    v8f z = {0.f,0.f,0.f,0.f,0.f,0.f,0.f,0.f};
    return z;
}
__device__ __forceinline__ v8f wmma16(v16h a, v16h b, v8f c) {
    return __builtin_amdgcn_wmma_f32_16x16x32_f16(false, a, false, b, (short)0, c, false, false);
}
// A fragment (16x32 f16 tile) from row-major f16 array with row stride lda (halves).
// ISA layout: lane<16: M=lane, halves 0..7 = K0..7, 8..15 = K16..23;
//             lane>=16: M=lane-16, halves 0..7 = K8..15, 8..15 = K24..31.
__device__ __forceinline__ v16h load_a(const _Float16* A, int lda, int mtile, int kt, int lane) {
    const _Float16* row = A + (size_t)(mtile * 16 + (lane & 15)) * lda + kt * 32 + (lane >> 4) * 8;
    v8h lo = *(const v8h*)(row);
    v8h hi = *(const v8h*)(row + 16);
    v16h a;
    #pragma unroll
    for (int i = 0; i < 8; ++i) { a[i] = lo[i]; a[i + 8] = hi[i]; }
    return a;
}
// B fragment: pre-packed, 16 contiguous halves per lane per (kt,nt) tile.
__device__ __forceinline__ v16h load_b(const _Float16* W, int ntiles, int kt, int nt, int lane) {
    return *(const v16h*)(W + ((((size_t)kt * ntiles + nt) * 32) + lane) * 16);
}
// CDNA5 async copy: 16 bytes global -> LDS, tracked by ASYNCcnt (ISA 15.18.3 op 98).
__device__ __forceinline__ void async_ld16(const void* g, void* lds) {
    asm volatile("global_load_async_to_lds_b128 %0, %1, off"
                 :: "v"((uint32_t)(uintptr_t)lds), "v"((uint64_t)(uintptr_t)g)
                 : "memory");
}
__device__ __forceinline__ void wait_async0() {
    asm volatile("s_wait_asynccnt 0" ::: "memory");
}

// ---------------- kernel 0: pack weights f32 -> f16 WMMA-B fragments ----------------
// src: Kact x N row-major f32.  dst: [Kp/32][N/16][32 lanes][16 halves], zero-padded K.
__global__ __launch_bounds__(256) void k_pack(const float* __restrict__ src, _Float16* __restrict__ dst,
                                              int Kact, int Kp, int N) {
    int total = (Kp >> 5) * (N >> 4) * 512;
    int ntiles = N >> 4;
    for (int idx = blockIdx.x * 256 + threadIdx.x; idx < total; idx += gridDim.x * 256) {
        int h    = idx & 15;
        int lane = (idx >> 4) & 31;
        int tile = idx >> 9;
        int nt = tile % ntiles, kt = tile / ntiles;
        int n = nt * 16 + (lane & 15);
        int k = kt * 32 + (h & 7) + ((h >> 3) << 4) + ((lane >> 4) << 3);
        float v = (k < Kact) ? src[(size_t)k * N + n] : 0.f;
        dst[idx] = (_Float16)v;
    }
}

// ---------------- kernel 1: q -> Qh, tq (16 batch rows per block) ----------------
__global__ __launch_bounds__(256) void k_batch(
    const float* __restrict__ QE, const float* __restrict__ b_qp,
    const float* __restrict__ b_aq, const float* __restrict__ b_tq,
    const _Float16* __restrict__ pw_qp, const _Float16* __restrict__ pw_aq,
    const _Float16* __restrict__ pw_tq,
    float* __restrict__ wsQh, _Float16* __restrict__ wsTq)
{
    __shared__ __align__(32) _Float16 Aqe[16][QD];
    __shared__ __align__(32) _Float16 Aq[16][HDIM];
    const int b0 = blockIdx.x * 16;
    const int t = threadIdx.x, lane = t & 31, wave = t >> 5;

    for (int i = t; i < 16 * QD; i += 256)
        Aqe[i >> 9][i & 511] = (_Float16)QE[(size_t)(b0 + (i >> 9)) * QD + (i & 511)];
    __syncthreads();

    for (int nt = wave; nt < 24; nt += 8) {
        v8f accq = vzero8(), acct = vzero8();
        for (int kt = 0; kt < 16; ++kt) {
            v16h a = load_a(&Aqe[0][0], QD, 0, kt, lane);
            accq = wmma16(a, load_b(pw_qp, 24, kt, nt, lane), accq);
            acct = wmma16(a, load_b(pw_tq, 24, kt, nt, lane), acct);
        }
        int n = nt * 16 + (lane & 15), mb = (lane < 16) ? 0 : 8;
        float bq = b_qp[n], bt = b_tq[n];
        #pragma unroll
        for (int i = 0; i < 8; ++i) {
            int m = mb + i;
            Aq[m][n] = (_Float16)geluf(accq[i] + bq);
            wsTq[(size_t)(b0 + m) * HDIM + n] = (_Float16)geluf(acct[i] + bt);
        }
    }
    __syncthreads();
    for (int nt = wave; nt < 24; nt += 8) {
        v8f acc = vzero8();
        for (int kt = 0; kt < 12; ++kt)
            acc = wmma16(load_a(&Aq[0][0], HDIM, 0, kt, lane), load_b(pw_aq, 24, kt, nt, lane), acc);
        int n = nt * 16 + (lane & 15), mb = (lane < 16) ? 0 : 8;
        float ba = b_aq[n];
        #pragma unroll
        for (int i = 0; i < 8; ++i)
            wsQh[(size_t)(b0 + mb + i) * HDIM + n] = acc[i] + ba;
    }
}

// ---------------- kernel 2: per-batch neighbor pipeline ----------------
__global__ __launch_bounds__(256) void k_neighbor(
    const int* __restrict__ qcode, const float* __restrict__ qobs, const float* __restrict__ qmask,
    const float* __restrict__ rshifts, const unsigned char* __restrict__ rsmask,
    const int* __restrict__ rcodes, const float* __restrict__ rdist,
    const unsigned char* __restrict__ rvalid,
    const float* __restrict__ rc_table, const float* __restrict__ E_res,
    const float* __restrict__ b_np, const float* __restrict__ b_ss1,
    const float* __restrict__ W_ss2, const float* __restrict__ b_ss2,
    const float* __restrict__ b_ak, const float* __restrict__ b_av,
    const _Float16* __restrict__ pw_np, const _Float16* __restrict__ pw_ss1,
    const _Float16* __restrict__ pw_ak, const _Float16* __restrict__ pw_av,
    const float* __restrict__ wsQh,
    _Float16* __restrict__ wsAttn, float* __restrict__ wsShifts,
    float* __restrict__ wsPstats, float* __restrict__ wsGstats, float* __restrict__ wsAny)
{
    __shared__ __align__(32) _Float16 nbrA[KNBR][32];
    __shared__ __align__(32) _Float16 sdA[KNBR][32];
    __shared__ __align__(32) _Float16 kvS[KNBR][HDIM];
    __shared__ __align__(32) _Float16 KhS[KNBR][HDIM];
    __shared__ __align__(32) _Float16 VhS[KNBR][HDIM];
    __shared__ __align__(32) _Float16 hss[KNBR][HD2];
    __shared__ float s_shift[KNBR][NS], s_smask[KNBR][NS];
    __shared__ float s_dist[KNBR], s_same[KNBR];
    __shared__ int   s_valid[KNBR];
    __shared__ float s_bias[KNBR][NH];
    __shared__ float s_scores[NH][KNBR];
    __shared__ float s_w[NH][KNBR];
    __shared__ float s_qh[HDIM];
    __shared__ float s_any;

    const int b = blockIdx.x;
    const int t = threadIdx.x, lane = t & 31, wave = t >> 5;

    for (int i = t; i < HDIM; i += 256) s_qh[i] = wsQh[(size_t)b * HDIM + i];

    if (t < KNBR) {
        const int k = t, idx = b * KNBR + k;
        int rcode = rcodes[idx];
        int qc = qcode[b];
        int ridx = min(max(rcode, 0), NRT - 1), qidx = min(max(qc, 0), NRT - 1);
        float d = rdist[idx];
        int vld = rvalid[idx] ? 1 : 0;
        float same = (rcode == qc) ? 1.f : 0.f;
        s_dist[k] = d; s_same[k] = same; s_valid[k] = vld;
        for (int s = 0; s < NS; ++s) {
            float rs = rshifts[(size_t)idx * NS + s];
            float rq = rc_table[qidx * NS + s], rr = rc_table[ridx * NS + s];
            float corr = rq + rs - rr;
            if (!(rq == rq) || !(rr == rr)) corr = rs;      // NaN fallback
            float smf = rsmask[(size_t)idx * NS + s] ? 1.f : 0.f;
            float sft = (smf != 0.f) ? corr : rs;
            s_shift[k][s] = sft; s_smask[k][s] = smf;
            nbrA[k][s]     = (_Float16)sft;
            nbrA[k][6 + s] = (_Float16)smf;
            float qm = qmask[b * NS + s];
            sdA[k][s]     = (_Float16)((qobs[b * NS + s] - sft) * qm);
            sdA[k][6 + s] = (_Float16)qm;
        }
        for (int j = 0; j < 16; ++j) nbrA[k][12 + j] = (_Float16)E_res[ridx * 16 + j];
        nbrA[k][28] = (_Float16)d; nbrA[k][29] = (_Float16)same;
        nbrA[k][30] = (_Float16)0.f; nbrA[k][31] = (_Float16)0.f;
        for (int j = 12; j < 32; ++j) sdA[k][j] = (_Float16)0.f;
    }
    __syncthreads();
    if (t == 0) {
        int a = 0;
        for (int k = 0; k < KNBR; ++k) a |= s_valid[k];
        s_any = a ? 1.f : 0.f; wsAny[b] = s_any;
    }
    if (t < KNBR * NS) wsShifts[(size_t)b * (KNBR * NS) + t] = s_shift[t / NS][t % NS];

    // kv = gelu(nbr @ W_np + b_np)  [48 tiles] ; hss = gelu(sdiff @ W_ss1 + b_ss1) [24 tiles]
    for (int idx = wave; idx < 72; idx += 8) {
        if (idx < 48) {
            int mt = idx / 24, nt = idx % 24;
            v8f acc = wmma16(load_a(&nbrA[0][0], 32, mt, 0, lane),
                             load_b(pw_np, 24, 0, nt, lane), vzero8());
            int n = nt * 16 + (lane & 15), mb = ((lane < 16) ? 0 : 8) + mt * 16;
            float bb = b_np[n];
            #pragma unroll
            for (int i = 0; i < 8; ++i) kvS[mb + i][n] = (_Float16)geluf(acc[i] + bb);
        } else {
            int j = idx - 48, mt = j / 12, nt = j % 12;
            v8f acc = wmma16(load_a(&sdA[0][0], 32, mt, 0, lane),
                             load_b(pw_ss1, 12, 0, nt, lane), vzero8());
            int n = nt * 16 + (lane & 15), mb = ((lane < 16) ? 0 : 8) + mt * 16;
            float bb = b_ss1[n];
            #pragma unroll
            for (int i = 0; i < 8; ++i) hss[mb + i][n] = (_Float16)geluf(acc[i] + bb);
        }
    }
    __syncthreads();

    // Kh = kv@W_ak + b_ak ; Vh = kv@W_av + b_av.
    // Wave specialization: wave -> (which, mtile, nt-half); A fragments cached in registers
    // once (12 x v16h = 96 VGPRs) and reused across 12 N-tiles -> only B streams from L2.
    {
        const int which = wave & 1, mt = (wave >> 1) & 1, ntBase = (wave >> 2) * 12;
        const _Float16* W = which ? pw_av : pw_ak;
        const float* bias = which ? b_av : b_ak;
        v16h aF[12];
        #pragma unroll
        for (int kt = 0; kt < 12; ++kt) aF[kt] = load_a(&kvS[0][0], HDIM, mt, kt, lane);
        const int n0 = (lane & 15), mb = ((lane < 16) ? 0 : 8) + mt * 16;
        for (int nt = ntBase; nt < ntBase + 12; ++nt) {
            v8f acc = vzero8();
            #pragma unroll
            for (int kt = 0; kt < 12; ++kt)
                acc = wmma16(aF[kt], load_b(W, 24, kt, nt, lane), acc);
            int n = nt * 16 + n0;
            float bb = bias[n];
            if (which) { for (int i = 0; i < 8; ++i) VhS[mb + i][n] = (_Float16)(acc[i] + bb); }
            else       { for (int i = 0; i < 8; ++i) KhS[mb + i][n] = (_Float16)(acc[i] + bb); }
        }
    }
    __syncthreads();

    const float eps = 1e-8f;
    if (t < 192) {                       // score bias: (32,192)@(192,6)
        int k = t / NH, s = t % NH;
        float acc = b_ss2[s];
        for (int j = 0; j < HD2; ++j) acc += (float)hss[k][j] * W_ss2[j * NH + s];
        s_bias[k][s] = acc;
    } else if (t < 192 + NS) {           // per_stats
        int s = t - 192;
        float cnt = 0, sum = 0, sd = 0, st = 0;
        for (int k = 0; k < KNBR; ++k) {
            float vm = (s_valid[k] && s_smask[k][s] != 0.f) ? 1.f : 0.f;
            cnt += vm; sum += vm * s_shift[k][s]; sd += vm * s_dist[k]; st += vm * s_same[k];
        }
        float mean = sum / (cnt + eps), var = 0;
        for (int k = 0; k < KNBR; ++k) {
            float vm = (s_valid[k] && s_smask[k][s] != 0.f) ? 1.f : 0.f;
            float dv = s_shift[k][s] - mean; var += vm * dv * dv;
        }
        var /= (cnt + eps);
        float varn = fminf(fmaxf(log1pf(var), 0.f), 5.f) * 0.2f;
        float* o = &wsPstats[((size_t)b * NS + s) * 4];
        o[0] = cnt / (32.f + eps); o[1] = varn; o[2] = sd / (cnt + eps); o[3] = st / (cnt + eps);
    } else if (t == 192 + NS) {          // gstats
        float nv = 0, ns = 0, dsum = 0, dmax = 0;
        for (int k = 0; k < KNBR; ++k) {
            float vf = s_valid[k] ? 1.f : 0.f;
            nv += vf; ns += vf * s_same[k];
            float dm = s_valid[k] ? s_dist[k] : 0.f;
            dsum += dm; dmax = fmaxf(dmax, dm);
        }
        float* o = &wsGstats[(size_t)b * 4];
        o[0] = dsum / (nv + eps); o[1] = dmax; o[2] = nv / 32.f; o[3] = ns / 32.f;
    }
    __syncthreads();

    if (t < NH * KNBR) {                 // scores
        int h = t / KNBR, k = t % KNBR;
        float acc = 0;
        for (int d = 0; d < 64; ++d) acc += s_qh[h * 64 + d] * (float)KhS[k][h * 64 + d];
        acc = acc * 0.125f + s_bias[k][h];
        bool ve = s_valid[k] || ((s_any == 0.f) && k == 0);
        s_scores[h][k] = ve ? acc : -10000.0f;
    }
    __syncthreads();
    if (t < NH) {                        // softmax per head
        float mx = -1e30f;
        for (int k = 0; k < KNBR; ++k) mx = fmaxf(mx, s_scores[t][k]);
        float den = 0;
        for (int k = 0; k < KNBR; ++k) { float e = expf(s_scores[t][k] - mx); s_w[t][k] = e; den += e; }
        float inv = 1.f / den;
        for (int k = 0; k < KNBR; ++k) s_w[t][k] *= inv;
    }
    __syncthreads();
    if (t < HDIM) {                      // attn_out
        int h = t >> 6;
        float acc = 0;
        for (int k = 0; k < KNBR; ++k) acc += s_w[h][k] * (float)VhS[k][t];
        wsAttn[(size_t)b * HDIM + t] = (_Float16)acc;
    }
}

// ---------------- kernel 3: ctx / swb / tc (16 batch rows per block) ----------------
__global__ __launch_bounds__(256) void k_epilogue(
    const int* __restrict__ qcode, const float* __restrict__ E_trust,
    const float* __restrict__ b_cp1, const float* __restrict__ b_cp2,
    const float* __restrict__ b_sw1, const float* __restrict__ b_sw2,
    const float* __restrict__ W_sw2, const float* __restrict__ b_tc,
    const float* __restrict__ fallback_ctx,
    const _Float16* __restrict__ pw_cp1, const _Float16* __restrict__ pw_cp2,
    const _Float16* __restrict__ pw_sw1, const _Float16* __restrict__ pw_tc,
    const _Float16* __restrict__ wsAttn, const _Float16* __restrict__ wsTq,
    const float* __restrict__ wsGstats, const float* __restrict__ wsAny,
    float* __restrict__ out_ctx, float* __restrict__ wsSwb, _Float16* __restrict__ wsTc)
{
    __shared__ __align__(32) _Float16 Aat[16][HDIM];
    __shared__ __align__(32) _Float16 H1[16][HDIM];
    __shared__ __align__(32) _Float16 H2[16][HDIM];
    __shared__ __align__(32) _Float16 Atc[16][448];
    const int b0 = blockIdx.x * 16;
    const int t = threadIdx.x, lane = t & 31, wave = t >> 5;

    // Async-DMA the contiguous f16 blocks into LDS (ASYNCcnt path), overlap with tail fill.
    for (int c = t; c < 768; c += 256)                       // Aat <- wsAttn (12 KB contiguous)
        async_ld16(wsAttn + (size_t)b0 * HDIM + c * 8, ((_Float16*)&Aat[0][0]) + c * 8);
    for (int c = t; c < 768; c += 256) {                     // Atc[:, 0:384] <- wsTq (row-wise)
        int row = c / 48, chunk = c % 48;
        async_ld16(wsTq + (size_t)(b0 + row) * HDIM + chunk * 8, &Atc[row][chunk * 8]);
    }
    for (int i = t; i < 16 * 64; i += 256) {                 // Atc[:, 384:448] tail
        int row = i >> 6, c = 384 + (i & 63), bb = b0 + row;
        _Float16 v;
        if (c < 416) { int qc = min(max(qcode[bb], 0), NRT - 1); v = (_Float16)E_trust[qc * 32 + (c - 384)]; }
        else if (c < 420) v = (_Float16)wsGstats[(size_t)bb * 4 + (c - 416)];
        else v = (_Float16)0.f;
        Atc[row][c] = v;
    }
    wait_async0();
    __syncthreads();

    if (wave < 4) {
        // waves 0..3: A = Aat (cached once), each wave 6 N-tiles, produce BOTH H1 (cp1) and H2 (sw1).
        v16h aF[12];
        #pragma unroll
        for (int kt = 0; kt < 12; ++kt) aF[kt] = load_a(&Aat[0][0], HDIM, 0, kt, lane);
        const int n0 = (lane & 15), mb = (lane < 16) ? 0 : 8;
        for (int nt = wave * 6; nt < wave * 6 + 6; ++nt) {
            v8f a1 = vzero8(), a2 = vzero8();
            #pragma unroll
            for (int kt = 0; kt < 12; ++kt) {
                a1 = wmma16(aF[kt], load_b(pw_cp1, 24, kt, nt, lane), a1);
                a2 = wmma16(aF[kt], load_b(pw_sw1, 24, kt, nt, lane), a2);
            }
            int n = nt * 16 + n0;
            float bb1 = b_cp1[n], bb2 = b_sw1[n];
            #pragma unroll
            for (int i = 0; i < 8; ++i) {
                H1[mb + i][n] = (_Float16)geluf(a1[i] + bb1);
                H2[mb + i][n] = (_Float16)geluf(a2[i] + bb2);
            }
        }
    } else {
        // waves 4..7: tc = gelu(Atc @ W_tc + b_tc), A cached (14 k-tiles), 6 N-tiles each.
        v16h aF[14];
        #pragma unroll
        for (int kt = 0; kt < 14; ++kt) aF[kt] = load_a(&Atc[0][0], 448, 0, kt, lane);
        const int n0 = (lane & 15), mb = (lane < 16) ? 0 : 8;
        for (int nt = (wave - 4) * 6; nt < (wave - 4) * 6 + 6; ++nt) {
            v8f acc = vzero8();
            #pragma unroll
            for (int kt = 0; kt < 14; ++kt)
                acc = wmma16(aF[kt], load_b(pw_tc, 24, kt, nt, lane), acc);
            int n = nt * 16 + n0;
            float bb = b_tc[n];
            for (int i = 0; i < 8; ++i)
                wsTc[(size_t)(b0 + mb + i) * HDIM + n] = (_Float16)geluf(acc[i] + bb);
        }
    }
    __syncthreads();

    {   // ctx = H1 @ W_cp2 + b  (+ fallback); A cached, 3 N-tiles per wave.
        v16h aF[12];
        #pragma unroll
        for (int kt = 0; kt < 12; ++kt) aF[kt] = load_a(&H1[0][0], HDIM, 0, kt, lane);
        const int n0 = (lane & 15), mb = (lane < 16) ? 0 : 8;
        for (int nt = wave; nt < 24; nt += 8) {
            v8f acc = vzero8();
            #pragma unroll
            for (int kt = 0; kt < 12; ++kt)
                acc = wmma16(aF[kt], load_b(pw_cp2, 24, kt, nt, lane), acc);
            int n = nt * 16 + n0;
            float bb = b_cp2[n], fb = fallback_ctx[n];
            #pragma unroll
            for (int i = 0; i < 8; ++i) {
                int m = mb + i;
                float v = acc[i] + bb;
                if (wsAny[b0 + m] == 0.f) v = fb;
                out_ctx[(size_t)(b0 + m) * HDIM + n] = v;
            }
        }
    }
    if (t < 16 * NS) {                        // swb = H2 @ W_sw2 + b_sw2
        int row = t / NS, s = t % NS;
        float acc = b_sw2[s];
        for (int j = 0; j < HDIM; ++j) acc += (float)H2[row][j] * W_sw2[j * NS + s];
        wsSwb[(size_t)(b0 + row) * NS + s] = acc;
    }
}

// ---------------- kernel 4: trust head (16 (b,s) rows per block) ----------------
__global__ __launch_bounds__(256) void k_trust(
    const _Float16* __restrict__ wsTc, const float* __restrict__ wsPstats,
    const float* __restrict__ b_tp1, const float* __restrict__ W_tp2,
    const float* __restrict__ b_tp2, const _Float16* __restrict__ pw_tp1,
    float* __restrict__ out_trust)
{
    __shared__ __align__(32) _Float16 A[16][416];
    __shared__ float Hd[16][HD2];
    const int r0 = blockIdx.x * 16;
    const int t = threadIdx.x, lane = t & 31, wave = t >> 5;

    for (int c = t; c < 16 * 48; c += 256) {                 // A[:, 0:384] <- wsTc via async DMA
        int row = c / 48, chunk = c % 48;
        int bb = (r0 + row) / NS;
        async_ld16(wsTc + (size_t)bb * HDIM + chunk * 8, &A[row][chunk * 8]);
    }
    for (int i = t; i < 16 * 32; i += 256) {                 // A[:, 384:416] tail
        int row = i >> 5, c = 384 + (i & 31);
        int r = r0 + row, bb = r / NS, s = r % NS;
        A[row][c] = (c < 388) ? (_Float16)wsPstats[((size_t)bb * NS + s) * 4 + (c - 384)]
                              : (_Float16)0.f;
    }
    wait_async0();
    __syncthreads();

    for (int nt = wave; nt < 12; nt += 8) {
        v8f acc = vzero8();
        for (int kt = 0; kt < 13; ++kt)
            acc = wmma16(load_a(&A[0][0], 416, 0, kt, lane), load_b(pw_tp1, 12, kt, nt, lane), acc);
        int n = nt * 16 + (lane & 15), mb = (lane < 16) ? 0 : 8;
        float bb = b_tp1[n];
        #pragma unroll
        for (int i = 0; i < 8; ++i) Hd[mb + i][n] = geluf(acc[i] + bb);
    }
    __syncthreads();
    if (t < 16) {
        float acc = b_tp2[0];
        for (int j = 0; j < HD2; ++j) acc += Hd[t][j] * W_tp2[j];
        out_trust[r0 + t] = 1.f / (1.f + expf(-acc));
    }
}

// ---------------- kernel 5: transferred shifts ----------------
__global__ __launch_bounds__(256) void k_transfer(
    const int* __restrict__ qcode, const int* __restrict__ rcodes,
    const float* __restrict__ rdist, const unsigned char* __restrict__ rvalid,
    const unsigned char* __restrict__ rsmask,
    const float* __restrict__ wsShifts, const float* __restrict__ wsSwb,
    const float* __restrict__ wsAny, const float* __restrict__ shift_scale,
    const float* __restrict__ fallback_shift, float* __restrict__ out_trans)
{
    int g = blockIdx.x * 256 + threadIdx.x;
    if (g >= BATCH * NS) return;
    int b = g / NS, s = g % NS;
    float sw = wsSwb[g] * 0.1f;
    int qc = qcode[b];
    float wt[KNBR], sf[KNBR];
    float mx = -1e30f;
    for (int k = 0; k < KNBR; ++k) {
        int idx = b * KNBR + k;
        bool vm = rvalid[idx] && rsmask[(size_t)idx * NS + s];
        float st = (rcodes[idx] == qc) ? 0.5f : 0.f;
        float w = rdist[idx] + st + sw;
        w = vm ? w : -10000.0f;
        wt[k] = w; sf[k] = wsShifts[(size_t)idx * NS + s];
        mx = fmaxf(mx, w);
    }
    float den = 0, num = 0;
    for (int k = 0; k < KNBR; ++k) {
        float e = expf(wt[k] - mx);
        den += e; num += e * sf[k];
    }
    float out = (num / den) * shift_scale[s];
    if (wsAny[b] == 0.f) out = fallback_shift[s];
    out_trans[g] = out;
}

// ---------------- host launcher ----------------
extern "C" void kernel_launch(void* const* d_in, const int* in_sizes, int n_in,
                              void* d_out, int out_size, void* d_ws, size_t ws_size,
                              hipStream_t stream) {
    (void)in_sizes; (void)n_in; (void)out_size; (void)ws_size;
    // inputs (setup_inputs order)
    const float* QE      = (const float*)d_in[0];
    const int*   qcode   = (const int*)d_in[1];
    const float* qobs    = (const float*)d_in[2];
    const float* qmask   = (const float*)d_in[3];
    const float* rshifts = (const float*)d_in[4];
    const unsigned char* rsmask = (const unsigned char*)d_in[5];
    const int*   rcodes  = (const int*)d_in[6];
    const float* rdist   = (const float*)d_in[7];
    const unsigned char* rvalid = (const unsigned char*)d_in[8];
    const float* rc_table = (const float*)d_in[9];
    const float* W_qp = (const float*)d_in[10]; const float* b_qp = (const float*)d_in[11];
    const float* E_res = (const float*)d_in[12];
    const float* W_np = (const float*)d_in[13]; const float* b_np = (const float*)d_in[14];
    const float* W_aq = (const float*)d_in[15]; const float* b_aq = (const float*)d_in[16];
    const float* W_ak = (const float*)d_in[17]; const float* b_ak = (const float*)d_in[18];
    const float* W_av = (const float*)d_in[19]; const float* b_av = (const float*)d_in[20];
    const float* W_ss1 = (const float*)d_in[21]; const float* b_ss1 = (const float*)d_in[22];
    const float* W_ss2 = (const float*)d_in[23]; const float* b_ss2 = (const float*)d_in[24];
    const float* W_cp1 = (const float*)d_in[25]; const float* b_cp1 = (const float*)d_in[26];
    const float* W_cp2 = (const float*)d_in[27]; const float* b_cp2 = (const float*)d_in[28];
    const float* W_sw1 = (const float*)d_in[29]; const float* b_sw1 = (const float*)d_in[30];
    const float* W_sw2 = (const float*)d_in[31]; const float* b_sw2 = (const float*)d_in[32];
    const float* E_trust = (const float*)d_in[33];
    const float* W_tq = (const float*)d_in[34]; const float* b_tq = (const float*)d_in[35];
    const float* W_tc = (const float*)d_in[36]; const float* b_tc = (const float*)d_in[37];
    const float* W_tp1 = (const float*)d_in[38]; const float* b_tp1 = (const float*)d_in[39];
    const float* W_tp2 = (const float*)d_in[40]; const float* b_tp2 = (const float*)d_in[41];
    const float* shift_scale = (const float*)d_in[42];
    const float* fallback_shift = (const float*)d_in[43];
    const float* fallback_ctx = (const float*)d_in[44];

    float* out_ctx   = (float*)d_out;                  // (B,H)
    float* out_trans = out_ctx + (size_t)BATCH * HDIM; // (B,S)
    float* out_trust = out_trans + (size_t)BATCH * NS; // (B,S)

    // workspace carve-out
    char* ws = (char*)d_ws;
    size_t off = 0;
    auto carve = [&](size_t bytes) -> char* {
        char* p = ws + off;
        off += (bytes + 255) & ~(size_t)255;
        return p;
    };
    auto fragN = [](int Kp, int N) -> size_t { return (size_t)(Kp / 32) * (N / 16) * 512; };
    _Float16* pw_qp  = (_Float16*)carve(fragN(512, 384) * 2);
    _Float16* pw_aq  = (_Float16*)carve(fragN(384, 384) * 2);
    _Float16* pw_ak  = (_Float16*)carve(fragN(384, 384) * 2);
    _Float16* pw_av  = (_Float16*)carve(fragN(384, 384) * 2);
    _Float16* pw_np  = (_Float16*)carve(fragN(32, 384) * 2);
    _Float16* pw_ss1 = (_Float16*)carve(fragN(32, 192) * 2);
    _Float16* pw_cp1 = (_Float16*)carve(fragN(384, 384) * 2);
    _Float16* pw_cp2 = (_Float16*)carve(fragN(384, 384) * 2);
    _Float16* pw_sw1 = (_Float16*)carve(fragN(384, 384) * 2);
    _Float16* pw_tq  = (_Float16*)carve(fragN(512, 384) * 2);
    _Float16* pw_tc  = (_Float16*)carve(fragN(448, 384) * 2);
    _Float16* pw_tp1 = (_Float16*)carve(fragN(416, 192) * 2);
    float*    wsQh     = (float*)carve((size_t)BATCH * HDIM * 4);
    _Float16* wsTq     = (_Float16*)carve((size_t)BATCH * HDIM * 2);
    _Float16* wsAttn   = (_Float16*)carve((size_t)BATCH * HDIM * 2);
    _Float16* wsTc     = (_Float16*)carve((size_t)BATCH * HDIM * 2);
    float*    wsShifts = (float*)carve((size_t)BATCH * KNBR * NS * 4);
    float*    wsPstats = (float*)carve((size_t)BATCH * NS * 4 * 4);
    float*    wsGstats = (float*)carve((size_t)BATCH * 4 * 4);
    float*    wsSwb    = (float*)carve((size_t)BATCH * NS * 4);
    float*    wsAny    = (float*)carve((size_t)BATCH * 4);

    // pack weights
    k_pack<<<256, 256, 0, stream>>>(W_qp,  pw_qp,  512, 512, 384);
    k_pack<<<256, 256, 0, stream>>>(W_aq,  pw_aq,  384, 384, 384);
    k_pack<<<256, 256, 0, stream>>>(W_ak,  pw_ak,  384, 384, 384);
    k_pack<<<256, 256, 0, stream>>>(W_av,  pw_av,  384, 384, 384);
    k_pack<<<256, 256, 0, stream>>>(W_np,  pw_np,   30,  32, 384);
    k_pack<<<256, 256, 0, stream>>>(W_ss1, pw_ss1,  12,  32, 192);
    k_pack<<<256, 256, 0, stream>>>(W_cp1, pw_cp1, 384, 384, 384);
    k_pack<<<256, 256, 0, stream>>>(W_cp2, pw_cp2, 384, 384, 384);
    k_pack<<<256, 256, 0, stream>>>(W_sw1, pw_sw1, 384, 384, 384);
    k_pack<<<256, 256, 0, stream>>>(W_tq,  pw_tq,  512, 512, 384);
    k_pack<<<256, 256, 0, stream>>>(W_tc,  pw_tc,  420, 448, 384);
    k_pack<<<256, 256, 0, stream>>>(W_tp1, pw_tp1, 388, 416, 192);

    k_batch<<<BATCH / 16, 256, 0, stream>>>(QE, b_qp, b_aq, b_tq, pw_qp, pw_aq, pw_tq, wsQh, wsTq);

    k_neighbor<<<BATCH, 256, 0, stream>>>(
        qcode, qobs, qmask, rshifts, rsmask, rcodes, rdist, rvalid,
        rc_table, E_res, b_np, b_ss1, W_ss2, b_ss2, b_ak, b_av,
        pw_np, pw_ss1, pw_ak, pw_av, wsQh,
        wsAttn, wsShifts, wsPstats, wsGstats, wsAny);

    k_epilogue<<<BATCH / 16, 256, 0, stream>>>(
        qcode, E_trust, b_cp1, b_cp2, b_sw1, b_sw2, W_sw2, b_tc, fallback_ctx,
        pw_cp1, pw_cp2, pw_sw1, pw_tc, wsAttn, wsTq, wsGstats, wsAny,
        out_ctx, wsSwb, wsTc);

    k_trust<<<(BATCH * NS) / 16, 256, 0, stream>>>(
        wsTc, wsPstats, b_tp1, W_tp2, b_tp2, pw_tp1, out_trust);

    k_transfer<<<(BATCH * NS + 255) / 256, 256, 0, stream>>>(
        qcode, rcodes, rdist, rvalid, rsmask, wsShifts, wsSwb, wsAny,
        shift_scale, fallback_shift, out_trans);
}
